// SelfAttention_2534030704870
// MI455X (gfx1250) — matrix operand order repaired and checked by reading, not compiled, not run on hardware
//
#include <hip/hip_runtime.h>
#include <hip/hip_bf16.h>
#include <math.h>

typedef __attribute__((ext_vector_type(16))) __bf16 v16bf;
typedef __attribute__((ext_vector_type(8)))  float  v8f;

union BFrag { v16bf v; unsigned u[8]; };

// hardware bf16 convert (v_cvt_pk_bf16_f32 path)
__device__ __forceinline__ unsigned short f2bf(float f) {
    union { __bf16 h; unsigned short s; } r;
    r.h = (__bf16)f;
    return r.s;
}
__device__ __forceinline__ unsigned pack2(float a, float b) {
    union { __bf16 h[2]; unsigned u; } r;
    r.h[0] = (__bf16)a;
    r.h[1] = (__bf16)b;
    return r.u;
}

// ---------------------------------------------------------------------------
// Kernel 1: fused QKV projection.
//   Y(320x4096) = [Wq;Wk;Wv](320x256) * x[b](256x4096) + bias, per batch.
// Combined weights staged once per WG into LDS as bf16 (160KB of the 320KB
// WGP LDS). Each wave owns 16 columns of x, builds its K=256 B-fragments
// once in registers, then sweeps all 20 row tiles:
// hot loop = 2x ds_load_b128 + 1x v_wmma_f32_16x16x32_bf16 per K-step.
// ---------------------------------------------------------------------------
__global__ __launch_bounds__(256) void qkv_proj_kernel(
    const float* __restrict__ x,
    const float* __restrict__ Wq, const float* __restrict__ bq,
    const float* __restrict__ Wk, const float* __restrict__ bk,
    const float* __restrict__ Wv, const float* __restrict__ bv,
    unsigned short* __restrict__ fw, unsigned short* __restrict__ gw,
    unsigned short* __restrict__ hw)
{
    __shared__ __align__(16) unsigned short w_lds[320 * 256]; // [m][c] bf16, 160KB

    const int tid  = threadIdx.x;
    const int lane = tid & 31;
    const int wave = tid >> 5;
    const int l15  = lane & 15;
    const bool hiL = lane >= 16;

    const int b      = blockIdx.x >> 5;        // 4 batches
    const int colblk = blockIdx.x & 31;        // 32 column blocks of 128
    const int n      = colblk * 128 + wave * 16 + l15;

    // ---- stage combined [Wq;Wk;Wv] into LDS as bf16 (pairs of elements) ----
    {
        unsigned* wl = (unsigned*)w_lds;
#pragma unroll 4
        for (int p = tid; p < 320 * 256 / 2; p += 256) {
            int e = p * 2;                      // flat element = m*256 + c
            const float* src;
            if (e < 32 * 256)      src = Wq + e;
            else if (e < 64 * 256) src = Wk + (e - 32 * 256);
            else                   src = Wv + (e - 64 * 256);
            float2 wp = *(const float2*)src;
            wl[p] = pack2(wp.x, wp.y);
        }
    }

    // ---- build persistent B fragments: x columns, full K=256 ----
    const float* xb = x + (size_t)b * 256 * 4096;
    BFrag xf[8];
#pragma unroll
    for (int kk = 0; kk < 8; ++kk) {
#pragma unroll
        for (int v = 0; v < 8; ++v) {
            int c = kk * 32 + (v < 4 ? 2 * v : 2 * v + 8) + (hiL ? 8 : 0);
            float x0 = xb[(size_t)c * 4096 + n];
            float x1 = xb[(size_t)(c + 1) * 4096 + n];
            xf[kk].u[v] = pack2(x0, x1);
        }
    }

    __syncthreads();

    // ---- sweep the 20 row tiles ----
#pragma unroll 1
    for (int rt = 0; rt < 20; ++rt) {
        const int row0 = rt * 16;               // uniform
        const int mA   = row0 + l15;            // A row for this lane

        v8f acc = {0.f, 0.f, 0.f, 0.f, 0.f, 0.f, 0.f, 0.f};
#pragma unroll
        for (int kk = 0; kk < 8; ++kk) {
            BFrag A;
#pragma unroll
            for (int v = 0; v < 8; ++v) {
                // A (16x32): lanes0-15 K=0..7,16..23 ; lanes16-31 +8
                int c = kk * 32 + (v < 4 ? 2 * v : 2 * v + 8) + (hiL ? 8 : 0);
                A.u[v] = *(const unsigned*)&w_lds[mA * 256 + c];
            }
            acc = __builtin_amdgcn_wmma_f32_16x16x32_bf16(
                false, A.v, false, xf[kk].v, (short)0, acc, false, false);
        }

        // ---- uniform class dispatch + bias + bf16 store ----
        unsigned short* dst; const float* bias; int cb; size_t rows;
        if (row0 < 32)      { dst = fw; bias = bq; cb = 0;  rows = 32;  }
        else if (row0 < 64) { dst = gw; bias = bk; cb = 32; rows = 32;  }
        else                { dst = hw; bias = bv; cb = 64; rows = 256; }
#pragma unroll
        for (int v = 0; v < 8; ++v) {
            int m = row0 + v + (hiL ? 8 : 0);   // D layout: M = vgpr (+8 hi half)
            int r = m - cb;
            dst[((size_t)b * rows + r) * 4096 + n] = f2bf(acc[v] + bias[r]);
        }
    }
}

// ---------------------------------------------------------------------------
// Kernel 2: fused flash attention + residual, double-buffered with
// GLOBAL_LOAD_ASYNC_TO_LDS_B128 (ASYNCcnt) for the 16KB h-tile staging.
// Per wave: 16-query tile. Loop over 32-key blocks:
//   S^T(32j x 16i) = g^T(j x c) * f(c x i)   -> 2x wmma bf16 (K=CQK=32)
//   online softmax (per-lane + shfl_xor(16) reductions)
//   O(256c x 16i) += h(c x j) * P^T(j x i)   -> 16x wmma bf16 (K=32 keys)
// Epilogue: out = gamma * O/l + x.
// ---------------------------------------------------------------------------
__global__ __launch_bounds__(256) void attn_kernel(
    const float* __restrict__ x,
    const unsigned short* __restrict__ fw,
    const unsigned short* __restrict__ gw,
    const unsigned short* __restrict__ hw,
    const float* __restrict__ gamma,
    float* __restrict__ out)
{
    __shared__ __align__(16) unsigned short h_lds[2][256 * 32]; // [buf][c][j] 2x16KB
    __shared__ __align__(16) unsigned short g_lds[2][32 * 32];  // [buf][j][c] 2x2KB

    const int tid  = threadIdx.x;
    const int lane = tid & 31;
    const int wave = tid >> 5;
    const int l15  = lane & 15;
    const bool hiL = lane >= 16;

    const int b    = blockIdx.x >> 5;          // 32 blocks per batch
    const int qblk = blockIdx.x & 31;          // 128 queries per block
    const int i    = qblk * 128 + wave * 16 + l15;

    const unsigned short* fb = fw + (size_t)b * 32 * 4096;
    const unsigned short* gb = gw + (size_t)b * 32 * 4096;
    const unsigned short* hb = hw + (size_t)b * 256 * 4096;

    // ---- async stage of one 256x32 h tile (64B per thread, 4x b128) ----
    auto stage_h_async = [&](int j0, int buf) {
        // low 32 bits of a generic shared pointer == LDS byte offset
        unsigned ldsoff = (unsigned)(unsigned long long)&h_lds[buf][tid * 32];
        const unsigned short* gp = hb + (size_t)tid * 4096 + j0;
        asm volatile(
            "global_load_async_to_lds_b128 %0, %1, off\n\t"
            "global_load_async_to_lds_b128 %0, %1, off offset:16\n\t"
            "global_load_async_to_lds_b128 %0, %1, off offset:32\n\t"
            "global_load_async_to_lds_b128 %0, %1, off offset:48"
            :: "v"(ldsoff), "v"(gp) : "memory");
    };
    // ---- synchronous transposed stage of one 32x32 g tile ----
    auto stage_g = [&](int j0, int buf) {
        int c  = tid >> 3;                  // 0..31
        int jj = (tid & 7) * 4;             // 0..28
        const unsigned short* src = gb + (size_t)c * 4096 + j0 + jj;
        ushort2 p0 = *(const ushort2*)(src);
        ushort2 p1 = *(const ushort2*)(src + 2);
        g_lds[buf][(jj + 0) * 32 + c] = p0.x;
        g_lds[buf][(jj + 1) * 32 + c] = p0.y;
        g_lds[buf][(jj + 2) * 32 + c] = p1.x;
        g_lds[buf][(jj + 3) * 32 + c] = p1.y;
    };

    // persistent B fragment: f (K=c 32, N=i 16); lanes0-15 c=0..15, hi c=16..31
    BFrag fB;
#pragma unroll
    for (int v = 0; v < 8; ++v) {
        int c = 2 * v + (hiL ? 16 : 0);
        unsigned lo  = fb[(size_t)c * 4096 + i];
        unsigned hi2 = fb[(size_t)(c + 1) * 4096 + i];
        fB.u[v] = lo | (hi2 << 16);
    }

    v8f accO[16];
#pragma unroll
    for (int t = 0; t < 16; ++t)
#pragma unroll
        for (int v = 0; v < 8; ++v) accO[t][v] = 0.f;

    float m_run = -INFINITY;
    float l_run = 0.f;

    // prologue: stage block 0
    stage_h_async(0, 0);
    stage_g(0, 0);

    int buf = 0;
#pragma unroll 1
    for (int j0 = 0; j0 < 4096; j0 += 32) {
        // own async writes landed; then barrier makes all threads' tiles visible
        asm volatile("s_wait_asynccnt 0" ::: "memory");
        __syncthreads();

        // issue next block into the other buffer (overlaps with compute below)
        if (j0 + 32 < 4096) {
            stage_h_async(j0 + 32, buf ^ 1);
            stage_g(j0 + 32, buf ^ 1);
        }

        const unsigned short* hT = h_lds[buf];
        const unsigned short* gT = g_lds[buf];

        // ---- S^T = g^T * f : two 16x16x32 WMMAs (j rows 0..15 and 16..31) ----
        BFrag gA0, gA1;
#pragma unroll
        for (int v = 0; v < 8; ++v) {
            int cb = 2 * v + (hiL ? 16 : 0);
            gA0.u[v] = *(const unsigned*)&gT[l15 * 32 + cb];
            gA1.u[v] = *(const unsigned*)&gT[(l15 + 16) * 32 + cb];
        }
        v8f z = {0.f, 0.f, 0.f, 0.f, 0.f, 0.f, 0.f, 0.f};
        v8f d0 = __builtin_amdgcn_wmma_f32_16x16x32_bf16(
            false, gA0.v, false, fB.v, (short)0, z, false, false);
        v8f d1 = __builtin_amdgcn_wmma_f32_16x16x32_bf16(
            false, gA1.v, false, fB.v, (short)0, z, false, false);

        // ---- online softmax over the 32 keys of this block ----
        float bmax = -INFINITY;
#pragma unroll
        for (int v = 0; v < 8; ++v) {
            bmax = fmaxf(bmax, d0[v]);
            bmax = fmaxf(bmax, d1[v]);
        }
        bmax = fmaxf(bmax, __shfl_xor(bmax, 16, 32));
        float m_new = fmaxf(m_run, bmax);
        float scale = __expf(m_run - m_new);
        float bsum = 0.f;
#pragma unroll
        for (int v = 0; v < 8; ++v) {
            d0[v] = __expf(d0[v] - m_new);
            d1[v] = __expf(d1[v] - m_new);
            bsum += d0[v] + d1[v];
        }
        bsum += __shfl_xor(bsum, 16, 32);
        l_run = l_run * scale + bsum;
        m_run = m_new;
#pragma unroll
        for (int t = 0; t < 16; ++t)
#pragma unroll
            for (int v = 0; v < 8; ++v) accO[t][v] *= scale;

        // ---- convert P^T (two f32 D tiles) -> one bf16 B fragment (K=j, N=i) ----
        float d0s[8], d1s[8];
#pragma unroll
        for (int v = 0; v < 8; ++v) {
            d0s[v] = __shfl_xor(d0[v], 16, 32);
            d1s[v] = __shfl_xor(d1[v], 16, 32);
        }
        BFrag pB;
#pragma unroll
        for (int v = 0; v < 8; ++v) {
            float a, c2;
            if (v < 4) { // lanes0-15: j=2v,2v+1 (own d0); hi lanes: j=16+2v (partner's d1)
                a  = hiL ? d1s[2 * v]     : d0[2 * v];
                c2 = hiL ? d1s[2 * v + 1] : d0[2 * v + 1];
            } else {     // lanes0-15: j in 8..15 (partner's d0); hi lanes: own d1
                a  = hiL ? d1[2 * v - 8]  : d0s[2 * v - 8];
                c2 = hiL ? d1[2 * v - 7]  : d0s[2 * v - 7];
            }
            pB.u[v] = pack2(a, c2);
        }

        // ---- O(c x i) += h(c x j) * P^T(j x i): 16 WMMAs over channel tiles ----
#pragma unroll
        for (int ct = 0; ct < 16; ++ct) {
            BFrag hA;
            int c = ct * 16 + l15;
#pragma unroll
            for (int v = 0; v < 4; ++v) {
                int jb = 4 * v + (hiL ? 16 : 0);
                uint2 p = *(const uint2*)&hT[c * 32 + jb];
                hA.u[2 * v]     = p.x;
                hA.u[2 * v + 1] = p.y;
            }
            accO[ct] = __builtin_amdgcn_wmma_f32_16x16x32_bf16(
                false, hA.v, false, pB.v, (short)0, accO[ct], false, false);
        }

        buf ^= 1;
    }

    // ---- epilogue: out = gamma * O/l + x ----
    const float inv_l = 1.0f / l_run;
    const float gm = gamma[0];
    const float* xb = x + (size_t)b * 256 * 4096;
    float* ob = out + (size_t)b * 256 * 4096;
#pragma unroll
    for (int ct = 0; ct < 16; ++ct) {
#pragma unroll
        for (int v = 0; v < 8; ++v) {
            int c = ct * 16 + v + (hiL ? 8 : 0);
            size_t idx = (size_t)c * 4096 + i;
            ob[idx] = gm * (accO[ct][v] * inv_l) + xb[idx];
        }
    }
}

// ---------------------------------------------------------------------------
extern "C" void kernel_launch(void* const* d_in, const int* in_sizes, int n_in,
                              void* d_out, int out_size, void* d_ws, size_t ws_size,
                              hipStream_t stream) {
    const float* x     = (const float*)d_in[0];
    const float* Wq    = (const float*)d_in[1];
    const float* bq    = (const float*)d_in[2];
    const float* Wk    = (const float*)d_in[3];
    const float* bk    = (const float*)d_in[4];
    const float* Wv    = (const float*)d_in[5];
    const float* bv    = (const float*)d_in[6];
    const float* gamma = (const float*)d_in[7];

    // workspace layout (bf16): f (4x32x4096), g (4x32x4096), h (4x256x4096) = 10MB
    unsigned short* fw = (unsigned short*)d_ws;
    unsigned short* gw = fw + (size_t)4 * 32 * 4096;
    unsigned short* hw = gw + (size_t)4 * 32 * 4096;

    // 4 batches x 32 column-blocks (128 cols each: 8 waves x 16)
    qkv_proj_kernel<<<128, 256, 0, stream>>>(x, Wq, bq, Wk, bk, Wv, bv, fw, gw, hw);
    // 4 batches x 32 query-blocks (128 queries each, 8 waves x 16)
    attn_kernel<<<128, 256, 0, stream>>>(x, fw, gw, hw, gamma, (float*)d_out);
}
